// BasicBlock_6571299963501
// MI455X (gfx1250) — 8-bit matrix operand order repaired and checked by reading, compiled, not run on hardware
//
#include <hip/hip_runtime.h>
#include <hip/hip_bf16.h>


#define NB   64
#define CH   256
#define HH   28
#define WW   28
#define HW   (HH*WW)
#define Q    (NB*HW)
#define PT   16
#define TAPS 9
#define CB   4
#define APACK_BYTES_PER_PTILE 36864
#define APACK_INTS_PER_PTILE  9216

typedef __attribute__((ext_vector_type(8))) int v8i;
typedef __attribute__((ext_vector_type(4))) int v4i;
typedef __attribute__((ext_vector_type(4))) unsigned int v4u;

__device__ __forceinline__ int chan_perm(int c) {
  return (c & ~0x30) | (((c >> 4) & 1) << 5) | (((c >> 5) & 1) << 4);
}

__global__ __launch_bounds__(256)
void stats_nchw(const float* __restrict__ x, const float* __restrict__ g,
                const float* __restrict__ b, float* __restrict__ scale,
                float* __restrict__ shift) {
  __shared__ float ssum[256], ssq[256];
  const int ch = blockIdx.x, t = threadIdx.x;
  float s = 0.f, s2 = 0.f;
  for (int i = t; i < Q; i += 256) {
    int n = i / HW, r = i - n * HW;
    float v = x[((size_t)n * CH + ch) * HW + r];
    s += v; s2 += v * v;
  }
  ssum[t] = s; ssq[t] = s2; __syncthreads();
  for (int o = 128; o > 0; o >>= 1) {
    if (t < o) { ssum[t] += ssum[t + o]; ssq[t] += ssq[t + o]; }
    __syncthreads();
  }
  if (t == 0) {
    float m = ssum[0] * (1.f / Q);
    float var = ssq[0] * (1.f / Q) - m * m;
    float sc = g[ch] * rsqrtf(var + 1e-5f);
    scale[ch] = sc; shift[ch] = b[ch] - m * sc;
  }
}

__global__ __launch_bounds__(256)
void stats_nhwc(const float* __restrict__ x, const float* __restrict__ g,
                const float* __restrict__ b, float* __restrict__ scale,
                float* __restrict__ shift) {
  __shared__ float ssum[256], ssq[256];
  const int ch = blockIdx.x, t = threadIdx.x;
  float s = 0.f, s2 = 0.f;
  for (int i = t; i < Q; i += 256) {
    float v = x[(size_t)i * CH + ch];
    s += v; s2 += v * v;
  }
  ssum[t] = s; ssq[t] = s2; __syncthreads();
  for (int o = 128; o > 0; o >>= 1) {
    if (t < o) { ssum[t] += ssum[t + o]; ssq[t] += ssq[t + o]; }
    __syncthreads();
  }
  if (t == 0) {
    float m = ssum[0] * (1.f / Q);
    float var = ssq[0] * (1.f / Q) - m * m;
    float sc = g[ch] * rsqrtf(var + 1e-5f);
    scale[ch] = sc; shift[ch] = b[ch] - m * sc;
  }
}

__global__ __launch_bounds__(256)
void binarize_nchw(const float* __restrict__ x, const float* __restrict__ scale,
                   const float* __restrict__ shift, signed char* __restrict__ xb) {
  const int idx = blockIdx.x * 256 + threadIdx.x;
  const int cpos = idx & 255, q = idx >> 8;
  const int c = chan_perm(cpos);
  const int n = q / HW, s = q - n * HW;
  const float v = x[((size_t)n * CH + c) * HW + s];
  const float t = v * scale[c] + shift[c];
  xb[idx] = (signed char)((t > 0.f) - (t < 0.f));
}

__global__ __launch_bounds__(256)
void binarize_nhwc(const float* __restrict__ x, const float* __restrict__ scale,
                   const float* __restrict__ shift, signed char* __restrict__ xb) {
  const int idx = blockIdx.x * 256 + threadIdx.x;
  const int cpos = idx & 255, q = idx >> 8;
  const int c = chan_perm(cpos);
  const float t = x[(size_t)q * CH + c] * scale[c] + shift[c];
  xb[idx] = (signed char)((t > 0.f) - (t < 0.f));
}

__global__ __launch_bounds__(256)
void wmean_kernel(const float* __restrict__ w, float* __restrict__ wmean) {
  const int id = blockIdx.x * 256 + threadIdx.x;
  if (id >= CH * TAPS) return;
  const int p = id / TAPS, tap = id - p * TAPS;
  float s = 0.f;
  for (int c = 0; c < CH; ++c) s += w[((size_t)p * CH + c) * TAPS + tap];
  wmean[id] = s * (1.f / CH);
}

__global__ __launch_bounds__(256)
void wpack_kernel(const float* __restrict__ w, const float* __restrict__ wmean,
                  signed char* __restrict__ apack) {
  const int idx = blockIdx.x * 256 + threadIdx.x;
  if (idx >= PT * APACK_BYTES_PER_PTILE) return;
  const int ptile = idx / APACK_BYTES_PER_PTILE;
  int rem = idx - ptile * APACK_BYTES_PER_PTILE;
  const int tap = rem / 4096; rem &= 4095;
  const int cb = rem >> 10;   rem &= 1023;
  const int lane = rem >> 5;
  const int j = rem & 31;
  const int g = lane >> 4, m = lane & 15;
  const int v = j >> 2, bb = j & 3;
  const int K = ((v & 4) << 3) | (g << 4) | ((v & 2) << 2) | ((v & 1) << 2) | bb;
  const int c = cb * 64 + K;
  const int p = ptile * 16 + m;
  const float wv = w[((size_t)p * CH + c) * TAPS + tap] - wmean[p * TAPS + tap];
  apack[idx] = (signed char)((wv > 0.f) - (wv < 0.f));
}

__global__ __launch_bounds__(256)
void binconv_wmma(const signed char* __restrict__ xb,
                  const signed char* __restrict__ apack,
                  const float* __restrict__ gamma, const float* __restrict__ alpha,
                  const float* __restrict__ beta, float* __restrict__ outp,
                  const int doRelu) {
  __shared__ int ldsA[APACK_INTS_PER_PTILE];
  const int ptile = blockIdx.y;
  const unsigned long long ga =
      (unsigned long long)(apack + (size_t)ptile * APACK_BYTES_PER_PTILE);

#if __has_builtin(__builtin_amdgcn_tensor_load_to_lds)
  if (threadIdx.x < 32) {
    v4u g0;
    g0[0] = 1u;
    g0[1] = (unsigned int)(unsigned long long)(uintptr_t)&ldsA[0];
    g0[2] = (unsigned int)ga;
    g0[3] = (unsigned int)((ga >> 32) & 0x01FFFFFFu) | (2u << 30);
    v8i g1;
    g1[0] = 0x30000;
    g1[1] = (int)(4608u << 16);
    g1[2] = (int)(1u << 16);
    g1[3] = (int)(4608u << 16);
    g1[4] = 1;
    g1[5] = 4608;
    g1[6] = (int)(4608u << 16);
    g1[7] = 0;
    v4i z4 = {0, 0, 0, 0};
#if defined(__clang_major__) && __clang_major__ >= 23
    v8i z8 = {0, 0, 0, 0, 0, 0, 0, 0};
    __builtin_amdgcn_tensor_load_to_lds(g0, g1, z4, z4, z8, 0);
#else
    __builtin_amdgcn_tensor_load_to_lds(g0, g1, z4, z4, 0);
#endif
    __builtin_amdgcn_s_wait_tensorcnt(0);
  }
#else
  {
    const v4i* src = (const v4i*)ga;
    v4i* dst = (v4i*)ldsA;
    for (int i = threadIdx.x; i < APACK_INTS_PER_PTILE / 4; i += 256) dst[i] = src[i];
  }
#endif
  __syncthreads();

  const int wave = threadIdx.x >> 5;
  const int lane = threadIdx.x & 31;
  const int col  = lane & 15;
  const int g32  = (lane >> 4) << 5;
  const int q    = (blockIdx.x * 8 + wave) * 16 + col;
  const int n = q / HW, s = q - n * HW;
  const int h = s / WW, w = s - h * WW;

  const v8i vzero = {0, 0, 0, 0, 0, 0, 0, 0};
  v8i acc = vzero;

  #pragma unroll
  for (int tap = 0; tap < TAPS; ++tap) {
    const int kh = tap / 3 - 1, kw = tap % 3 - 1;
    const int hh = h + kh, ww2 = w + kw;
    const bool valid = (hh >= 0) & (hh < HH) & (ww2 >= 0) & (ww2 < WW);
    const signed char* bptr = xb + (size_t)(n * HW + hh * WW + ww2) * CH + g32;
    if (valid) __builtin_prefetch(bptr, 0, 1);
    #pragma unroll
    for (int cb = 0; cb < CB; ++cb) {
      v8i bmat = vzero;
      if (valid) bmat = *(const v8i*)(bptr + cb * 64);
      const v8i amat = *(const v8i*)&ldsA[(tap * CB + cb) * 256 + lane * 8];
      acc = __builtin_amdgcn_wmma_i32_16x16x64_iu8(true, amat, true, bmat, acc,
                                                   false, false);
    }
  }

  const float ab_sc = alpha[h] * beta[w];
  const int g8 = (lane >> 4) << 3;
  #pragma unroll
  for (int r = 0; r < 8; ++r) {
    const int p = ptile * 16 + r + g8;
    float v = (float)acc[r] * gamma[p] * ab_sc;
    if (doRelu) v = fmaxf(v, 0.f);
    outp[(size_t)q * CH + p] = v;
  }
}

__global__ __launch_bounds__(256)
void final_kernel(const float* __restrict__ c2, const float* __restrict__ scale,
                  const float* __restrict__ shift, const float* __restrict__ x,
                  float* __restrict__ out) {
  const int idx = blockIdx.x * 256 + threadIdx.x;
  const int n = idx / (CH * HW);
  int rem = idx - n * (CH * HW);
  const int p = rem / HW;
  const int s = rem - p * HW;
  const int q = n * HW + s;
  const float v = c2[(size_t)q * CH + p] * scale[p] + shift[p] + x[idx];
  out[idx] = fmaxf(v, 0.f);
}

extern "C" void kernel_launch(void* const* d_in, const int* in_sizes, int n_in,
                              void* d_out, int out_size, void* d_ws, size_t ws_size,
                              hipStream_t stream) {
  const float* x      = (const float*)d_in[0];
  const float* bn1_g  = (const float*)d_in[1];
  const float* bn1_b  = (const float*)d_in[2];
  const float* bn2_g  = (const float*)d_in[3];
  const float* bn2_b  = (const float*)d_in[4];
  const float* bn3_g  = (const float*)d_in[5];
  const float* bn3_b  = (const float*)d_in[6];
  const float* w1     = (const float*)d_in[7];
  const float* gamma1 = (const float*)d_in[8];
  const float* alpha1 = (const float*)d_in[9];
  const float* beta1  = (const float*)d_in[10];
  const float* w2     = (const float*)d_in[11];
  const float* gamma2 = (const float*)d_in[12];
  const float* alpha2 = (const float*)d_in[13];
  const float* beta2  = (const float*)d_in[14];
  float* out = (float*)d_out;

  char* ws = (char*)d_ws;
  auto carve = [&](size_t n) { char* p = ws; ws += (n + 255) & ~(size_t)255; return p; };
  float* bn1sc = (float*)carve(CH * 4);
  float* bn1sh = (float*)carve(CH * 4);
  float* bn2sc = (float*)carve(CH * 4);
  float* bn2sh = (float*)carve(CH * 4);
  float* bn3sc = (float*)carve(CH * 4);
  float* bn3sh = (float*)carve(CH * 4);
  float* wmean1 = (float*)carve(CH * TAPS * 4);
  float* wmean2 = (float*)carve(CH * TAPS * 4);
  signed char* apack1 = (signed char*)carve(PT * APACK_BYTES_PER_PTILE);
  signed char* apack2 = (signed char*)carve(PT * APACK_BYTES_PER_PTILE);
  signed char* xb1 = (signed char*)carve((size_t)Q * CH);
  signed char* xb2 = (signed char*)carve((size_t)Q * CH);
  float* f1 = (float*)carve((size_t)Q * CH * 4);

  const dim3 blk(256);
  const dim3 convGrid(Q / (16 * 8), PT);

  stats_nchw<<<CH, blk, 0, stream>>>(x, bn1_g, bn1_b, bn1sc, bn1sh);
  binarize_nchw<<<Q, blk, 0, stream>>>(x, bn1sc, bn1sh, xb1);
  wmean_kernel<<<TAPS, blk, 0, stream>>>(w1, wmean1);
  wmean_kernel<<<TAPS, blk, 0, stream>>>(w2, wmean2);
  wpack_kernel<<<2304, blk, 0, stream>>>(w1, wmean1, apack1);
  wpack_kernel<<<2304, blk, 0, stream>>>(w2, wmean2, apack2);
  binconv_wmma<<<convGrid, blk, 0, stream>>>(xb1, apack1, gamma1,
                                             alpha1, beta1, f1, 1);
  stats_nhwc<<<CH, blk, 0, stream>>>(f1, bn2_g, bn2_b, bn2sc, bn2sh);
  binarize_nhwc<<<Q, blk, 0, stream>>>(f1, bn2sc, bn2sh, xb2);
  binconv_wmma<<<convGrid, blk, 0, stream>>>(xb2, apack2, gamma2,
                                             alpha2, beta2, f1, 0);
  stats_nhwc<<<CH, blk, 0, stream>>>(f1, bn3_g, bn3_b, bn3sc, bn3sh);
  final_kernel<<<NB * CH * HW / 256, blk, 0, stream>>>(f1, bn3sc, bn3sh, x, out);
}
